// QNet_3710851743897
// MI455X (gfx1250) — matrix operand order repaired and checked by reading, not compiled, not run on hardware
//
#include <hip/hip_runtime.h>
#include <hip/hip_bf16.h>

typedef __attribute__((ext_vector_type(16))) __bf16 v16bf;
typedef __attribute__((ext_vector_type(8)))  __bf16 v8bf;
typedef __attribute__((ext_vector_type(8)))  float  v8f;
typedef __attribute__((ext_vector_type(4)))  float  v4f;

#define STR1 264   // 256 + 8 pad (bf16 elems)
#define STR2 136   // 128 + 8 pad
#define STR3 72    // 64 + 8 pad

// ---------------------------------------------------------------- init mu bits
__global__ void k_init_mu(unsigned* muIn, unsigned* muNot) {
    muIn[threadIdx.x & 127] = 0u;
    muNot[threadIdx.x & 127] = 0u;
}

// ------------------------------------------------- masked per-column max (bits)
// masked-out entries are 0.0 in the reference, so max >= 0 always; positive
// float bit patterns order identically to unsigned ints -> uint atomicMax.
__global__ void k_masked_max(const float* __restrict__ enc,
                             const int* __restrict__ cs,
                             unsigned* __restrict__ muIn,
                             unsigned* __restrict__ muNot, int N) {
    __shared__ unsigned sIn[128], sNot[128];
    int tid = threadIdx.x;           // 256 threads
    int c = tid & 127;
    int rOff = tid >> 7;             // 0 or 1
    if (tid < 128) { sIn[tid] = 0u; sNot[tid] = 0u; }
    __syncthreads();
    unsigned mIn = 0u, mNot = 0u;
    for (long row = blockIdx.x * 2 + rOff; row < N; row += (long)gridDim.x * 2) {
        float v = enc[row * 128 + c];
        if (v > 0.0f) {
            unsigned b = __float_as_uint(v);
            if (cs[row] > 0) { if (b > mIn) mIn = b; }
            else             { if (b > mNot) mNot = b; }
        }
    }
    if (mIn)  atomicMax(&sIn[c], mIn);
    if (mNot) atomicMax(&sNot[c], mNot);
    __syncthreads();
    if (tid < 128) {
        if (sIn[tid])  atomicMax(&muIn[tid],  sIn[tid]);
        if (sNot[tid]) atomicMax(&muNot[tid], sNot[tid]);
    }
}

// ---------------------------------------------- fold mu terms into layer-1 bias
__global__ void k_bias1(const float* __restrict__ W1, const float* __restrict__ b1,
                        const float* __restrict__ mu /* [0:128)=in, [128:256)=not */,
                        float* __restrict__ bias1p) {
    int j = threadIdx.x;             // 256 threads, 1 block
    float s = b1[j];
    #pragma unroll 4
    for (int k = 0; k < 128; ++k)
        s += mu[k] * W1[k * 256 + j] + mu[128 + k] * W1[(128 + k) * 256 + j];
    bias1p[j] = s;
}

// ------------------------------------- pack weights to bf16 WMMA B-tile layout
// B tile (32K x 16N): lane L holds column n0+(L%16); element e holds K = k0+8*(L/16)+e (e<8), k0+16+8*(L/16)+(e-8) (e>=8).
// Stored lane-contiguous: tile*512 + lane*16 + e  -> main kernel loads 32B/lane.
__global__ void k_pack(const float* __restrict__ W1, const float* __restrict__ W2,
                       const float* __restrict__ W3, __bf16* __restrict__ packed) {
    int t = blockIdx.x;              // 144 tiles total
    int lane = threadIdx.x;          // 32 threads
    const float* Wm; int NT, ncols, rowOff, tt;
    if (t < 64)       { Wm = W1; tt = t;       NT = 16; ncols = 256; rowOff = 256; } // W1c = W1[256:384]
    else if (t < 128) { Wm = W2; tt = t - 64;  NT = 8;  ncols = 128; rowOff = 0;   }
    else              { Wm = W3; tt = t - 128; NT = 4;  ncols = 64;  rowOff = 0;   }
    int kt = tt / NT, nt = tt % NT;
    int k0 = kt * 32, n0 = nt * 16;
    int g = lane >> 4, n = lane & 15;
    __bf16* out = packed + (long)t * 512 + lane * 16;
    #pragma unroll
    for (int e = 0; e < 16; ++e) {
        int k = k0 + (e < 8 ? 8 * g + e : 16 + 8 * g + (e - 8));
        out[e] = (__bf16)Wm[(long)(rowOff + k) * ncols + n0 + n];
    }
}

// ------------------------------------------------------- fused MLP main kernel
__global__ __launch_bounds__(128) void k_main(
    const float* __restrict__ enc, const float* __restrict__ bias1p,
    const float* __restrict__ b2, const float* __restrict__ b3,
    const float* __restrict__ b4,
    const __bf16* __restrict__ w1p, const __bf16* __restrict__ w2p,
    const __bf16* __restrict__ w3p, const float* __restrict__ W4,
    float* __restrict__ logits, int N) {
    __shared__ __attribute__((aligned(32))) __bf16 sh1[4][16 * STR1];
    __shared__ __attribute__((aligned(32))) __bf16 sh2[4][16 * STR2];
    __shared__ __attribute__((aligned(32))) __bf16 sh3[4][16 * STR3];

    int lane = threadIdx.x & 31;
    int wv = threadIdx.x >> 5;       // 4 waves/block, 16 rows each
    int m = lane & 15, g = lane >> 4;
    long rowBase = (long)blockIdx.x * 64 + wv * 16;
    long row = rowBase + m;
    if (row >= N) row = N - 1;       // branchless tail: recompute, discard on store
    const float* rp = enc + row * 128;

    __bf16* h1 = sh1[wv];
    __bf16* h2 = sh2[wv];
    __bf16* h3 = sh3[wv];

    // ---- load 4 encoder A-tiles (16x32 bf16 each), WMMA A layout
    v16bf A[4];
    #pragma unroll
    for (int kt = 0; kt < 4; ++kt) {
        const float* p = rp + kt * 32 + 8 * g;
        v4f f0 = *(const v4f*)(p);
        v4f f1 = *(const v4f*)(p + 4);
        v4f f2 = *(const v4f*)(p + 16);
        v4f f3 = *(const v4f*)(p + 20);
        v16bf a;
        #pragma unroll
        for (int i = 0; i < 4; ++i) {
            a[i]      = (__bf16)f0[i];
            a[4 + i]  = (__bf16)f1[i];
            a[8 + i]  = (__bf16)f2[i];
            a[12 + i] = (__bf16)f3[i];
        }
        A[kt] = a;
    }

    // ---- layer 1: [16x128] @ [128x256] + b1'  -> leaky_relu -> LDS (bf16)
    #pragma unroll
    for (int nt = 0; nt < 16; ++nt) {
        v8f acc = {};
        #pragma unroll
        for (int kt = 0; kt < 4; ++kt) {
            v16bf b = *(const v16bf*)(w1p + (long)(kt * 16 + nt) * 512 + lane * 16);
            acc = __builtin_amdgcn_wmma_f32_16x16x32_bf16(
                false, A[kt], false, b, (short)0, acc, false, false);
        }
        float bias = bias1p[nt * 16 + m];
        #pragma unroll
        for (int r = 0; r < 8; ++r) {
            float v = acc[r] + bias;
            v = v > 0.0f ? v : 0.01f * v;
            h1[(r + 8 * g) * STR1 + nt * 16 + m] = (__bf16)v;
        }
    }
    __syncthreads();

    // ---- layer 2: [16x256] @ [256x128]
    v16bf HA[8];
    #pragma unroll
    for (int kt = 0; kt < 8; ++kt) {
        const __bf16* p = h1 + m * STR1 + kt * 32 + 8 * g;
        v8bf s0 = *(const v8bf*)(p);
        v8bf s1 = *(const v8bf*)(p + 16);
        v16bf a;
        #pragma unroll
        for (int i = 0; i < 8; ++i) { a[i] = s0[i]; a[8 + i] = s1[i]; }
        HA[kt] = a;
    }
    #pragma unroll
    for (int nt = 0; nt < 8; ++nt) {
        v8f acc = {};
        #pragma unroll
        for (int kt = 0; kt < 8; ++kt) {
            v16bf b = *(const v16bf*)(w2p + (long)(kt * 8 + nt) * 512 + lane * 16);
            acc = __builtin_amdgcn_wmma_f32_16x16x32_bf16(
                false, HA[kt], false, b, (short)0, acc, false, false);
        }
        float bias = b2[nt * 16 + m];
        #pragma unroll
        for (int r = 0; r < 8; ++r) {
            float v = acc[r] + bias;
            v = v > 0.0f ? v : 0.01f * v;
            h2[(r + 8 * g) * STR2 + nt * 16 + m] = (__bf16)v;
        }
    }
    __syncthreads();

    // ---- layer 3: [16x128] @ [128x64]
    v16bf GA[4];
    #pragma unroll
    for (int kt = 0; kt < 4; ++kt) {
        const __bf16* p = h2 + m * STR2 + kt * 32 + 8 * g;
        v8bf s0 = *(const v8bf*)(p);
        v8bf s1 = *(const v8bf*)(p + 16);
        v16bf a;
        #pragma unroll
        for (int i = 0; i < 8; ++i) { a[i] = s0[i]; a[8 + i] = s1[i]; }
        GA[kt] = a;
    }
    #pragma unroll
    for (int nt = 0; nt < 4; ++nt) {
        v8f acc = {};
        #pragma unroll
        for (int kt = 0; kt < 4; ++kt) {
            v16bf b = *(const v16bf*)(w3p + (long)(kt * 4 + nt) * 512 + lane * 16);
            acc = __builtin_amdgcn_wmma_f32_16x16x32_bf16(
                false, GA[kt], false, b, (short)0, acc, false, false);
        }
        float bias = b3[nt * 16 + m];
        #pragma unroll
        for (int r = 0; r < 8; ++r) {
            float v = acc[r] + bias;
            v = v > 0.0f ? v : 0.01f * v;
            h3[(r + 8 * g) * STR3 + nt * 16 + m] = (__bf16)v;
        }
    }
    __syncthreads();

    // ---- layer 4: [16x64] @ [64x1] (scalar dot, lanes 0-15)
    if (g == 0) {
        long orow = rowBase + m;
        if (orow < N) {
            float s = b4[0];
            #pragma unroll
            for (int k = 0; k < 64; ++k)
                s += (float)h3[m * STR3 + k] * W4[k];
            logits[orow] = s;
        }
    }
}

// ----------------------------------------------------------- softmax (3 stages)
__global__ void k_pmax(const float* __restrict__ logits, float* __restrict__ pmax, int N) {
    __shared__ float s[256];
    float mx = -3.4e38f;
    for (long i = (long)blockIdx.x * 256 + threadIdx.x; i < N; i += (long)gridDim.x * 256)
        mx = fmaxf(mx, logits[i]);
    s[threadIdx.x] = mx; __syncthreads();
    for (int o = 128; o > 0; o >>= 1) {
        if (threadIdx.x < o) s[threadIdx.x] = fmaxf(s[threadIdx.x], s[threadIdx.x + o]);
        __syncthreads();
    }
    if (threadIdx.x == 0) pmax[blockIdx.x] = s[0];
}

__global__ void k_rmax(const float* __restrict__ pmax, float* __restrict__ gmax) {
    __shared__ float s[256];
    s[threadIdx.x] = pmax[threadIdx.x]; __syncthreads();
    for (int o = 128; o > 0; o >>= 1) {
        if (threadIdx.x < o) s[threadIdx.x] = fmaxf(s[threadIdx.x], s[threadIdx.x + o]);
        __syncthreads();
    }
    if (threadIdx.x == 0) gmax[0] = s[0];
}

__global__ void k_psum(const float* __restrict__ logits, const float* __restrict__ gmax,
                       float* __restrict__ psum, int N) {
    __shared__ float s[256];
    float M = gmax[0];
    float acc = 0.0f;
    for (long i = (long)blockIdx.x * 256 + threadIdx.x; i < N; i += (long)gridDim.x * 256)
        acc += __expf(logits[i] - M);
    s[threadIdx.x] = acc; __syncthreads();
    for (int o = 128; o > 0; o >>= 1) {
        if (threadIdx.x < o) s[threadIdx.x] += s[threadIdx.x + o];
        __syncthreads();
    }
    if (threadIdx.x == 0) psum[blockIdx.x] = s[0];
}

__global__ void k_rsum(const float* __restrict__ psum, float* __restrict__ ginv) {
    __shared__ float s[256];
    s[threadIdx.x] = psum[threadIdx.x]; __syncthreads();
    for (int o = 128; o > 0; o >>= 1) {
        if (threadIdx.x < o) s[threadIdx.x] += s[threadIdx.x + o];
        __syncthreads();
    }
    if (threadIdx.x == 0) ginv[0] = 1.0f / s[0];
}

__global__ void k_out(const float* __restrict__ logits, const float* __restrict__ gmax,
                      const float* __restrict__ ginv, float* __restrict__ out, int N) {
    long i = (long)blockIdx.x * 256 + threadIdx.x;
    if (i < N) out[i] = __expf(logits[i] - gmax[0]) * ginv[0];
}

// --------------------------------------------------------------------- launch
extern "C" void kernel_launch(void* const* d_in, const int* in_sizes, int n_in,
                              void* d_out, int out_size, void* d_ws, size_t ws_size,
                              hipStream_t stream) {
    const float* enc = (const float*)d_in[0];
    const int*   cs  = (const int*)d_in[1];
    const float* W1  = (const float*)d_in[2];
    const float* b1  = (const float*)d_in[3];
    const float* W2  = (const float*)d_in[4];
    const float* b2  = (const float*)d_in[5];
    const float* W3  = (const float*)d_in[6];
    const float* b3  = (const float*)d_in[7];
    const float* W4  = (const float*)d_in[8];
    const float* b4  = (const float*)d_in[9];
    int N = in_sizes[0] / 128;

    float* W = (float*)d_ws;
    unsigned* muIn  = (unsigned*)W;          // 128
    unsigned* muNot = muIn + 128;            // 128
    float* bias1p   = W + 256;               // 256
    float* logits   = W + 512;               // N
    float* pmax     = logits + N;            // 256
    float* gmax     = pmax + 256;            // 1
    float* psum     = gmax + 1;              // 256
    float* ginv     = psum + 256;            // 1
    size_t packOffF = ((size_t)(512 + N + 514) + 7) & ~(size_t)7;  // 32B aligned
    __bf16* packed  = (__bf16*)(W + packOffF);                      // 144*512 bf16
    __bf16* w1p = packed;
    __bf16* w2p = packed + 64 * 512;
    __bf16* w3p = w2p + 64 * 512;

    k_init_mu<<<1, 256, 0, stream>>>(muIn, muNot);
    k_masked_max<<<1024, 256, 0, stream>>>(enc, cs, muIn, muNot, N);
    k_bias1<<<1, 256, 0, stream>>>(W1, b1, (const float*)muIn, bias1p);
    k_pack<<<144, 32, 0, stream>>>(W1, W2, W3, packed);
    k_main<<<(N + 63) / 64, 128, 0, stream>>>(enc, bias1p, b2, b3, b4,
                                              w1p, w2p, w3p, W4, logits, N);
    k_pmax<<<256, 256, 0, stream>>>(logits, pmax, N);
    k_rmax<<<1, 256, 0, stream>>>(pmax, gmax);
    k_psum<<<256, 256, 0, stream>>>(logits, gmax, psum, N);
    k_rsum<<<1, 256, 0, stream>>>(psum, ginv);
    k_out<<<(N + 255) / 256, 256, 0, stream>>>(logits, gmax, ginv, (float*)d_out, N);
}
